// LearnedVectorQuantizer_58488864637012
// MI455X (gfx1250) — matrix operand order repaired and checked by reading, not compiled, not run on hardware
//
#include <hip/hip_runtime.h>
#include <hip/hip_bf16.h>
#include <stdint.h>

// ---------------------------------------------------------------------------
// Learned VQ (EMA codebook) for MI455X / gfx1250, wave32 + WMMA + TDM.
//
// Outputs (concatenated flat in d_out, reference return order):
//   codes      uint8  [B,16]          (B*16 bytes)
//   recon      f32    [B,1024]
//   new_cb     f32    [16,256,64]
//   new_count  f32    [16,256]
//   new_weight f32    [16,256,64]
//
// Workspace: cb_hi (bf16), cb_lo (bf16), cnorm = 0.5*||c||^2 (f32)  ~1.06 MB
// ---------------------------------------------------------------------------

#define EMBED_DIM 1024
#define N_BOOKS   16
#define KENT      256
#define DDIM      64
#define CB_ELEMS  (N_BOOKS * KENT * DDIM)   // 262144
#define DECAY_F        0.99f
#define ONE_M_DECAY_F  0.01f
#define EPS_F          1e-5f

typedef __attribute__((ext_vector_type(16))) __bf16        v16bf;
typedef __attribute__((ext_vector_type(8)))  float         v8f;
typedef __attribute__((ext_vector_type(4)))  unsigned int  v4u;
typedef __attribute__((ext_vector_type(8)))  int           v8i;
typedef __attribute__((ext_vector_type(4)))  int           v4i;

union Frag16 { v4u u[2]; v16bf b; };

static __device__ __forceinline__ unsigned short f32_to_bf16_rne(float f) {
  unsigned int u = __float_as_uint(f);
  u += 0x7FFFu + ((u >> 16) & 1u);          // round-to-nearest-even
  return (unsigned short)(u >> 16);
}
static __device__ __forceinline__ float bf16u_to_f32(unsigned short h) {
  return __uint_as_float(((unsigned int)h) << 16);
}

#if __has_builtin(__builtin_amdgcn_tensor_load_to_lds)
#define VQ_HAVE_TDM 1
// 1D TDM copy of nelem bf16 elements global->LDS with +4-dword padding after
// every 32 dwords (turns 128B rows into 144B LDS stride = 72 ushorts).
static __device__ __forceinline__ void tdm_load_bf16_padded(
    const void* gsrc, unsigned lds_byte_addr, unsigned nelem) {
  unsigned long long ga = (unsigned long long)(uintptr_t)gsrc;
  v4u g0;
  g0[0] = 1u;                                            // count=1, user D#
  g0[1] = lds_byte_addr;                                 // bits 63:32 lds_addr
  g0[2] = (unsigned)ga;                                  // global_addr lo
  g0[3] = (unsigned)((ga >> 32) & 0x01FFFFFFu)           // global_addr[56:32]
        | 0x80000000u;                                   // type=2 ("image")
  v8i g1;
  g1[0] = (int)((1u << 16)      // data_size = 2 bytes
              | (1u << 20)      // pad_enable
              | (4u << 22)      // pad_interval: 2^(4+1)=32 dwords (128 B)
              | (3u << 25));    // pad_amount: 3 -> 4 dwords (16 B)
  g1[1] = (int)((nelem & 0xFFFFu) << 16);                // tensor_dim0 lo16
  g1[2] = (int)(((nelem >> 16) & 0xFFFFu) | (1u << 16)); // dim0 hi16, dim1=1
  g1[3] = (int)((nelem & 0xFFFFu) << 16);                // tile_dim0
  g1[4] = 0;                                             // tile_dim1/2 unused
  g1[5] = (int)nelem;                                    // dim0_stride lo32
  g1[6] = (int)((nelem & 0xFFFFu) << 16);                // dim1_stride lo16
  g1[7] = 0;
  v4i z4 = {};
#if defined(__clang_major__) && (__clang_major__ >= 23)
  v8i z8 = {};
  __builtin_amdgcn_tensor_load_to_lds(g0, g1, z4, z4, z8, 0);
#else
  __builtin_amdgcn_tensor_load_to_lds(g0, g1, z4, z4, 0);
#endif
}
#endif

// ---------------------------------------------------------------------------
// K0a: split codebooks into bf16 hi/lo; init out_weight = 0.99*ema_weight,
//      out_count = 0.99*ema_count.  (1024 blocks x 256)
// ---------------------------------------------------------------------------
__global__ __launch_bounds__(256) void vq_prep_kernel(
    const float* __restrict__ codebooks,
    const float* __restrict__ ema_count,
    const float* __restrict__ ema_weight,
    float* __restrict__ out_count,
    float* __restrict__ out_weight,
    unsigned short* __restrict__ cb_hi,
    unsigned short* __restrict__ cb_lo)
{
  int i = blockIdx.x * 256 + threadIdx.x;
  if (i < CB_ELEMS) {
    float c = codebooks[i];
    unsigned short h = f32_to_bf16_rne(c);
    float resid = c - bf16u_to_f32(h);
    cb_hi[i] = h;
    cb_lo[i] = f32_to_bf16_rne(resid);
    out_weight[i] = DECAY_F * ema_weight[i];
  }
  if (i < N_BOOKS * KENT) {
    out_count[i] = DECAY_F * ema_count[i];
  }
}

// ---------------------------------------------------------------------------
// K0b: cnorm[e] = 0.5 * ||codebooks[e,:]||^2   (16 blocks x 256)
// ---------------------------------------------------------------------------
__global__ __launch_bounds__(256) void vq_cnorm_kernel(
    const float* __restrict__ codebooks,
    float* __restrict__ cnorm)
{
  int e = blockIdx.x * 256 + threadIdx.x;
  if (e < N_BOOKS * KENT) {
    const float4* p = (const float4*)(codebooks + (size_t)e * DDIM);
    float s = 0.0f;
#pragma unroll
    for (int i = 0; i < DDIM / 4; ++i) {
      float4 v = p[i];
      s += v.x * v.x + v.y * v.y + v.z * v.z + v.w * v.w;
    }
    cnorm[e] = 0.5f * s;
  }
}

// ---------------------------------------------------------------------------
// K1: codes via WMMA distance GEMM + argmax reduction + EMA scatter.
// grid = (B/128, N_BOOKS), block = 256 threads = 8 waves.
// The book's bf16 codebook (hi+lo, 64KB) is DMA'd into LDS by the Tensor Data
// Mover (with D# padding giving a bank-conflict-free 144B row stride) while
// the block stages its x slice as split-bf16. Each wave then computes a
// 16x256 score tile with 96 v_wmma_f32_16x16x32_bf16, reduces argmax over
// lanes, and the block scatters 0.01*x / 0.01 with global float atomics.
// ---------------------------------------------------------------------------
__global__ __launch_bounds__(256) void vq_codes_scatter_kernel(
    const float* __restrict__ x,
    const unsigned short* __restrict__ cb_hi,
    const unsigned short* __restrict__ cb_lo,
    const float* __restrict__ cnorm,
    uint8_t* __restrict__ out_codes,
    float* __restrict__ out_count,
    float* __restrict__ out_weight)
{
  const int book    = blockIdx.y;
  const int rowbase = blockIdx.x * 128;
  const int tid     = threadIdx.x;

  // padded stride 72 ushorts (144 B = 36 dwords) -> conflict-free b128 loads
  __shared__ unsigned short lds_hi[128 * 72];
  __shared__ unsigned short lds_lo[128 * 72];
  __shared__ unsigned short cbh_lds[KENT * 72];
  __shared__ unsigned short cbl_lds[KENT * 72];
  __shared__ float cn_lds[KENT];
  __shared__ int   code_lds[128];

  const unsigned short* cbh_g = cb_hi + (size_t)book * KENT * DDIM;
  const unsigned short* cbl_g = cb_lo + (size_t)book * KENT * DDIM;

#ifdef VQ_HAVE_TDM
  // ---- kick off the codebook DMA first so it overlaps the x staging below
  if (tid < 32) {
    tdm_load_bf16_padded(cbh_g, (unsigned)(uintptr_t)&cbh_lds[0], KENT * DDIM);
    tdm_load_bf16_padded(cbl_g, (unsigned)(uintptr_t)&cbl_lds[0], KENT * DDIM);
  }
#else
  // fallback: cooperative b128 copy with explicit padding
#pragma unroll
  for (int k = 0; k < 16; ++k) {
    int c = tid + k * 256;              // 4096 v4u chunks of cb data
    int e = c >> 3, p = c & 7;
    *(v4u*)&cbh_lds[e * 72 + p * 8] = *(const v4u*)(cbh_g + e * 64 + p * 8);
    *(v4u*)&cbl_lds[e * 72 + p * 8] = *(const v4u*)(cbl_g + e * 64 + p * 8);
  }
#endif

  // ---- stage this block's x slice [128 rows x 64 cols] as bf16 hi/lo
  cn_lds[tid] = cnorm[book * KENT + tid];
#pragma unroll
  for (int k = 0; k < 8; ++k) {
    int f4  = tid + k * 256;            // 0..2047 float4s
    int row = f4 >> 4;                  // 0..127
    int d4  = (f4 & 15) << 2;           // 0..60
    const float4 v = *(const float4*)(x + (size_t)(rowbase + row) * EMBED_DIM
                                        + book * DDIM + d4);
    unsigned short h0 = f32_to_bf16_rne(v.x), h1 = f32_to_bf16_rne(v.y);
    unsigned short h2 = f32_to_bf16_rne(v.z), h3 = f32_to_bf16_rne(v.w);
    unsigned short l0 = f32_to_bf16_rne(v.x - bf16u_to_f32(h0));
    unsigned short l1 = f32_to_bf16_rne(v.y - bf16u_to_f32(h1));
    unsigned short l2 = f32_to_bf16_rne(v.z - bf16u_to_f32(h2));
    unsigned short l3 = f32_to_bf16_rne(v.w - bf16u_to_f32(h3));
    int base = row * 72 + d4;
    *(unsigned int*)&lds_hi[base + 0] = (unsigned int)h0 | ((unsigned int)h1 << 16);
    *(unsigned int*)&lds_hi[base + 2] = (unsigned int)h2 | ((unsigned int)h3 << 16);
    *(unsigned int*)&lds_lo[base + 0] = (unsigned int)l0 | ((unsigned int)l1 << 16);
    *(unsigned int*)&lds_lo[base + 2] = (unsigned int)l2 | ((unsigned int)l3 << 16);
  }
#ifdef VQ_HAVE_TDM
  if (tid < 32) __builtin_amdgcn_s_wait_tensorcnt(0);
#endif
  __syncthreads();

  // ---- WMMA score tiles + argmax
  const int wave   = tid >> 5;          // 0..7 -> 16-row tile
  const int lane   = tid & 31;
  const int lanelo = lane & 15;         // N within tile / M row select
  const int half   = lane >> 4;         // K-octet select (A), K-run select (B)
  const int m0     = wave * 16;

  // A fragments (kept live across all 16 column tiles)
  Frag16 Ahi[2], Alo[2];
  {
    const int mrow = m0 + lanelo;
#pragma unroll
    for (int s = 0; s < 2; ++s) {
      int kb = s * 32 + half * 8;       // ISA 16-bit A layout: two 8-elem runs
      Ahi[s].u[0] = *(const v4u*)&lds_hi[mrow * 72 + kb];
      Ahi[s].u[1] = *(const v4u*)&lds_hi[mrow * 72 + kb + 16];
      Alo[s].u[0] = *(const v4u*)&lds_lo[mrow * 72 + kb];
      Alo[s].u[1] = *(const v4u*)&lds_lo[mrow * 72 + kb + 16];
    }
  }

  float best[8];
  int   bidx[8];
#pragma unroll
  for (int r = 0; r < 8; ++r) { best[r] = -3.4e38f; bidx[r] = 0; }

  for (int t = 0; t < 16; ++t) {        // 16 column tiles of 16 entries
    const int ent = (t * 16 + lanelo) * 72
                  + half * 8;           // 16-bit B layout matching A: two 8-elem runs
    Frag16 Bh[2], Bl[2];
#pragma unroll
    for (int s = 0; s < 2; ++s) {
      Bh[s].u[0] = *(const v4u*)&cbh_lds[ent + s * 32];
      Bh[s].u[1] = *(const v4u*)&cbh_lds[ent + s * 32 + 16];
      Bl[s].u[0] = *(const v4u*)&cbl_lds[ent + s * 32];
      Bl[s].u[1] = *(const v4u*)&cbl_lds[ent + s * 32 + 16];
    }
    v8f acc = {};
#pragma unroll
    for (int s = 0; s < 2; ++s) {
      acc = __builtin_amdgcn_wmma_f32_16x16x32_bf16(false, Ahi[s].b, false, Bh[s].b,
                                                    (short)0, acc, false, false);
      acc = __builtin_amdgcn_wmma_f32_16x16x32_bf16(false, Alo[s].b, false, Bh[s].b,
                                                    (short)0, acc, false, false);
      acc = __builtin_amdgcn_wmma_f32_16x16x32_bf16(false, Ahi[s].b, false, Bl[s].b,
                                                    (short)0, acc, false, false);
    }
    const float cn = cn_lds[t * 16 + lanelo];
    const int kk = t * 16 + lanelo;
#pragma unroll
    for (int r = 0; r < 8; ++r) {
      float s = acc[r] - cn;            // argmax(x.c - 0.5||c||^2) == argmin d2
      if (s > best[r]) { best[r] = s; bidx[r] = kk; }
    }
  }

  // reduce argmax over the 16-lane N dimension (stays within half-wave)
#pragma unroll
  for (int m = 1; m <= 8; m <<= 1) {
#pragma unroll
    for (int r = 0; r < 8; ++r) {
      float ov = __shfl_xor(best[r], m, 32);
      int   oi = __shfl_xor(bidx[r], m, 32);
      if (ov > best[r] || (ov == best[r] && oi < bidx[r])) {
        best[r] = ov; bidx[r] = oi;
      }
    }
  }

  if (lanelo == 0) {                    // lanes 0 & 16: rows m0..m0+7 / +8..+15
#pragma unroll
    for (int r = 0; r < 8; ++r) {
      int row = m0 + half * 8 + r;
      code_lds[row] = bidx[r];
      out_codes[(size_t)(rowbase + row) * N_BOOKS + book] = (uint8_t)bidx[r];
    }
  }
  __syncthreads();

  // ---- EMA scatter (counts + weights) via global float atomics
  if (tid < 128) {
    atomicAdd(out_count + book * KENT + code_lds[tid], ONE_M_DECAY_F);
  }
#pragma unroll
  for (int k = 0; k < 32; ++k) {
    int idx = tid + k * 256;            // 0..8191 = 128 rows x 64 dims
    int row = idx >> 6;
    int d   = idx & 63;
    float xv = bf16u_to_f32(lds_hi[row * 72 + d]) + bf16u_to_f32(lds_lo[row * 72 + d]);
    atomicAdd(out_weight + ((size_t)(book * KENT + code_lds[row])) * DDIM + d,
              ONE_M_DECAY_F * xv);
  }
}

// ---------------------------------------------------------------------------
// K2: new_cb = new_weight / (new_count + eps)   (1024 blocks x 256)
// ---------------------------------------------------------------------------
__global__ __launch_bounds__(256) void vq_finalize_kernel(
    const float* __restrict__ out_weight,
    const float* __restrict__ out_count,
    float* __restrict__ out_cb)
{
  int i = blockIdx.x * 256 + threadIdx.x;
  if (i < CB_ELEMS) {
    out_cb[i] = out_weight[i] / (out_count[i >> 6] + EPS_F);
  }
}

// ---------------------------------------------------------------------------
// K3: recon[b] = new_cb[n, codes[b,n]]   (one block per row, float4 streams)
// ---------------------------------------------------------------------------
__global__ __launch_bounds__(256) void vq_recon_kernel(
    const uint8_t* __restrict__ codes,
    const float* __restrict__ cb,
    float* __restrict__ recon)
{
  const int b = blockIdx.x;
  const int q = threadIdx.x;            // 0..255 float4s of the row
  const int n = q >> 4;
  const int d4 = q & 15;
  const int code = codes[(size_t)b * N_BOOKS + n];
  float4 v = ((const float4*)cb)[(size_t)(n * KENT + code) * (DDIM / 4) + d4];
  ((float4*)recon)[(size_t)b * (EMBED_DIM / 4) + q] = v;
}

// ---------------------------------------------------------------------------
extern "C" void kernel_launch(void* const* d_in, const int* in_sizes, int n_in,
                              void* d_out, int out_size, void* d_ws, size_t ws_size,
                              hipStream_t stream) {
  const float* x          = (const float*)d_in[0];
  const float* codebooks  = (const float*)d_in[1];
  const float* ema_count  = (const float*)d_in[2];
  const float* ema_weight = (const float*)d_in[3];
  const int B = in_sizes[0] / EMBED_DIM;   // 32768

  // outputs, flat in return order (codes are uint8 per the reference)
  uint8_t* out_codes = (uint8_t*)d_out;
  float* out_recon  = (float*)(out_codes + (size_t)B * N_BOOKS);
  float* out_cb     = out_recon + (size_t)B * EMBED_DIM;
  float* out_count  = out_cb + CB_ELEMS;
  float* out_weight = out_count + N_BOOKS * KENT;

  // workspace: bf16 split codebooks + 0.5*||c||^2
  unsigned short* cb_hi = (unsigned short*)d_ws;
  unsigned short* cb_lo = cb_hi + CB_ELEMS;
  float* cnorm = (float*)(cb_lo + CB_ELEMS);

  vq_prep_kernel<<<CB_ELEMS / 256, 256, 0, stream>>>(
      codebooks, ema_count, ema_weight, out_count, out_weight, cb_hi, cb_lo);
  vq_cnorm_kernel<<<(N_BOOKS * KENT) / 256, 256, 0, stream>>>(codebooks, cnorm);
  vq_codes_scatter_kernel<<<dim3(B / 128, N_BOOKS), 256, 0, stream>>>(
      x, cb_hi, cb_lo, cnorm, out_codes, out_count, out_weight);
  vq_finalize_kernel<<<CB_ELEMS / 256, 256, 0, stream>>>(out_weight, out_count, out_cb);
  vq_recon_kernel<<<B, 256, 0, stream>>>(out_codes, out_cb, out_recon);
}
